// MultiheadSelfAttention_61529701482744
// MI455X (gfx1250) — matrix operand order repaired and checked by reading, not compiled, not run on hardware
//
#include <hip/hip_runtime.h>
#include <hip/hip_bf16.h>

typedef __attribute__((ext_vector_type(16))) __bf16 v16bf;
typedef __attribute__((ext_vector_type(8)))  __bf16 v8bf;
typedef __attribute__((ext_vector_type(8)))  float  v8f;

#define S_LEN  1024
#define M_LEN  1024
#define T_LEN  2048
#define BATCH  4
#define DMODEL 1024
#define NHEAD  16
#define DHEAD  64
#define E3     3072   // 3*H*DH

__device__ __forceinline__ v8f wmma_bf16(v16bf a, v16bf b, v8f c) {
  return __builtin_amdgcn_wmma_f32_16x16x32_bf16(false, a, false, b, (short)0,
                                                 c, false, false);
}

// A-operand (16x32 bf16): lanes 0-15 M=0..15 K{0..15}; lanes 16-31
// same M, K{16..31}
__device__ __forceinline__ v16bf load_a16(const __bf16* base, int ld,
                                          int row0, int k0, int l16, int hv) {
  const __bf16* p = base + (long)(row0 + l16) * ld + k0 + hv * 16;
  v8bf lo = *(const v8bf*)p;
  v8bf hi = *(const v8bf*)(p + 8);
  return __builtin_shufflevector(lo, hi, 0, 1, 2, 3, 4, 5, 6, 7,
                                 8, 9, 10, 11, 12, 13, 14, 15);
}

// ---------------------------------------------------------------------------
// Prep: x += pos_emb (fp32 copy for residual), bf16 concat [T,B,D]
// ---------------------------------------------------------------------------
__global__ void prep_kernel(const float* __restrict__ x,
                            const float* __restrict__ pos,
                            const float* __restrict__ mem,
                            float* __restrict__ xp,
                            __bf16* __restrict__ c) {
  int idx = blockIdx.x * blockDim.x + threadIdx.x;
  if (idx >= T_LEN * BATCH * DMODEL) return;
  int t = idx / (BATCH * DMODEL);
  int r = idx - t * (BATCH * DMODEL);
  int d = r & (DMODEL - 1);
  float v;
  if (t < M_LEN) {
    v = mem[idx];
  } else {
    int s = t - M_LEN;
    v = x[s * BATCH * DMODEL + r] + pos[s * DMODEL + d];
    xp[s * BATCH * DMODEL + r] = v;
  }
  c[idx] = (__bf16)v;
}

// ---------------------------------------------------------------------------
// Transpose+cast weights: fp32 [K][N] -> bf16 [N][K]
// ---------------------------------------------------------------------------
__global__ void tcast_kernel(const float* __restrict__ in,
                             __bf16* __restrict__ out, int K, int N) {
  int idx = blockIdx.x * blockDim.x + threadIdx.x;
  if (idx >= K * N) return;
  int n = idx / K;
  int k = idx - n * K;
  out[idx] = (__bf16)in[(long)k * N + n];
}

// ---------------------------------------------------------------------------
// Staged bf16 GEMM core: WG = 4 waves, wave tile 32(M)x64(N), WG tile 128x64.
// B tile (64n x 32k) async-loaded to double-buffered LDS, shared by all waves.
// ---------------------------------------------------------------------------
#define GEMM_CORE(A_, BT_, m0_, n0_, acc_)                                    \
  __shared__ __align__(16) __bf16 wlds[2][64 * 32];                           \
  int tid = threadIdx.x;                                                      \
  int lane = tid & 31, w = tid >> 5;                                          \
  int l16 = lane & 15, hv = lane >> 4;                                        \
  int m0_ = blockIdx.x * 128 + w * 32;                                        \
  int n0_ = blockIdx.y * 64;                                                  \
  v8f acc_[2][4] = {};                                                        \
  {                                                                           \
    /* stage k0 = 0 into buffer 0 */                                          \
    _Pragma("unroll") for (int i = 0; i < 2; ++i) {                           \
      int cc = tid + i * 128;                                                 \
      int nn = cc >> 2, ko = (cc & 3) * 8;                                    \
      const __bf16* g = BT_ + (long)(n0_ + nn) * DMODEL + ko;                 \
      unsigned lofs =                                                         \
          (unsigned)(unsigned long long)(const void*)&wlds[0][nn * 32 + ko];  \
      asm volatile("global_load_async_to_lds_b128 %0, %1, off" ::"v"(lofs),   \
                   "v"(g)                                                     \
                   : "memory");                                               \
    }                                                                         \
    asm volatile("s_wait_asynccnt 0" ::: "memory");                           \
    __syncthreads();                                                          \
  }                                                                           \
  for (int k0 = 0; k0 < DMODEL; k0 += 32) {                                   \
    int buf = (k0 >> 5) & 1;                                                  \
    if (k0 + 32 < DMODEL) {                                                   \
      _Pragma("unroll") for (int i = 0; i < 2; ++i) {                         \
        int cc = tid + i * 128;                                               \
        int nn = cc >> 2, ko = (cc & 3) * 8;                                  \
        const __bf16* g = BT_ + (long)(n0_ + nn) * DMODEL + k0 + 32 + ko;     \
        unsigned lofs = (unsigned)(unsigned long long)(                       \
            const void*)&wlds[buf ^ 1][nn * 32 + ko];                         \
        asm volatile("global_load_async_to_lds_b128 %0, %1, off" ::"v"(lofs), \
                     "v"(g)                                                   \
                     : "memory");                                             \
      }                                                                       \
      __builtin_prefetch(A_ + (long)(m0_ + l16) * DMODEL + k0 + 32, 0, 3);    \
      __builtin_prefetch(A_ + (long)(m0_ + 16 + l16) * DMODEL + k0 + 32, 0,   \
                         3);                                                  \
    }                                                                         \
    v16bf a0 = load_a16(A_, DMODEL, m0_, k0, l16, hv);                        \
    v16bf a1 = load_a16(A_, DMODEL, m0_ + 16, k0, l16, hv);                   \
    _Pragma("unroll") for (int j = 0; j < 4; ++j) {                           \
      v16bf b = *(const v16bf*)&wlds[buf][(j * 16 + l16) * 32 + hv * 16];     \
      acc_[0][j] = wmma_bf16(a0, b, acc_[0][j]);                              \
      acc_[1][j] = wmma_bf16(a1, b, acc_[1][j]);                              \
    }                                                                         \
    asm volatile("s_wait_asynccnt 0" ::: "memory");                           \
    __syncthreads();                                                          \
  }

// QKV GEMM: c[8192][1024] @ wT[3072][1024] -> qkv bf16 [8192][3072]
__global__ void __launch_bounds__(128)
qkv_gemm_kernel(const __bf16* __restrict__ A, const __bf16* __restrict__ BT,
                __bf16* __restrict__ C) {
  GEMM_CORE(A, BT, m0, n0, acc)
#pragma unroll
  for (int q = 0; q < 2; ++q)
#pragma unroll
    for (int j = 0; j < 4; ++j)
#pragma unroll
      for (int r = 0; r < 8; ++r)
        C[(long)(m0 + q * 16 + r + hv * 8) * E3 + n0 + j * 16 + l16] =
            (__bf16)acc[q][j][r];
}

// Output projection + residual: Y = xp + vec @ w_o (fp32)
__global__ void __launch_bounds__(128)
out_gemm_kernel(const __bf16* __restrict__ A, const __bf16* __restrict__ BT,
                const float* __restrict__ xp, float* __restrict__ Y) {
  GEMM_CORE(A, BT, m0, n0, acc)
#pragma unroll
  for (int q = 0; q < 2; ++q)
#pragma unroll
    for (int j = 0; j < 4; ++j)
#pragma unroll
      for (int r = 0; r < 8; ++r) {
        long row = m0 + q * 16 + r + hv * 8;
        int col = n0 + j * 16 + l16;
        Y[row * DMODEL + col] = xp[row * DMODEL + col] + acc[q][j][r];
      }
}

// ---------------------------------------------------------------------------
// V transpose: qkv V part -> vt[b][h][d][t] (t contiguous)
// ---------------------------------------------------------------------------
__global__ void vtrans_kernel(const __bf16* __restrict__ qkv,
                              __bf16* __restrict__ vt) {
  long idx = blockIdx.x * (long)blockDim.x + threadIdx.x;
  if (idx >= (long)BATCH * NHEAD * DHEAD * T_LEN) return;
  int  t  = (int)(idx & (T_LEN - 1));
  long r  = idx >> 11;
  int  d  = (int)(r & (DHEAD - 1));
  long r2 = r >> 6;
  int  h  = (int)(r2 & (NHEAD - 1));
  int  b  = (int)(r2 >> 4);
  vt[idx] = qkv[((long)t * BATCH + b) * E3 + 2 * DMODEL + h * DHEAD + d];
}

// ---------------------------------------------------------------------------
// Flash attention: 1 wave = one (b,h) x 32-query tile (2 WMMA M tiles).
// K-block = 32 keys; K/V B-operands shared across both query tiles.
// ---------------------------------------------------------------------------
__global__ void __launch_bounds__(128)
attn_kernel(const __bf16* __restrict__ qkv, const __bf16* __restrict__ vt,
            __bf16* __restrict__ vec) {
  __shared__ __align__(32) __bf16 plds[4][2 * 16 * 32];
  int lane = threadIdx.x & 31, w = threadIdx.x >> 5;
  int l16 = lane & 15, hv = lane >> 4;

  int task = blockIdx.x * 4 + w;        // 2048 wave tasks
  int qt = task & 31;                   // 32 query tiles of 32
  int bh = task >> 5;
  int h = bh & (NHEAD - 1);
  int b = bh >> 4;
  int q0 = qt * 32;

  // Q A-operands: 2 query subtiles x 2 head-dim chunks
  v16bf aq[2][2];
#pragma unroll
  for (int qi = 0; qi < 2; ++qi) {
    const __bf16* qrow =
        qkv + ((long)(M_LEN + q0 + qi * 16 + l16) * BATCH + b) * E3 +
        h * DHEAD;
#pragma unroll
    for (int c = 0; c < 2; ++c) {
      v8bf lo = *(const v8bf*)(qrow + c * 32 + hv * 16);
      v8bf hi = *(const v8bf*)(qrow + c * 32 + 8 + hv * 16);
      aq[qi][c] = __builtin_shufflevector(lo, hi, 0, 1, 2, 3, 4, 5, 6, 7,
                                          8, 9, 10, 11, 12, 13, 14, 15);
    }
  }

  const __bf16* kbase = qkv + (long)b * E3 + DMODEL + h * DHEAD;
  const __bf16* vbase = vt + (long)(b * NHEAD + h) * DHEAD * T_LEN;

  float mrow[2][8], lrow[2][8];
  v8f o[2][4] = {};
#pragma unroll
  for (int qi = 0; qi < 2; ++qi)
#pragma unroll
    for (int r = 0; r < 8; ++r) {
      mrow[qi][r] = -3.0e38f;
      lrow[qi][r] = 0.f;
    }

  const float cs = 0.125f * 1.44269504088896340736f;  // scale * log2(e)
  int kend = M_LEN + q0 + 32;
  if (kend > T_LEN) kend = T_LEN;

  for (int kb = 0; kb < kend; kb += 32) {
    // ---- scores for both query tiles; K operands loaded once ----
    v8f s[2][2] = {};
#pragma unroll
    for (int c = 0; c < 2; ++c) {
      const __bf16* krow = kbase + (long)(kb + c * 16 + l16) * (BATCH * E3);
      v16bf bk0 = *(const v16bf*)(krow + hv * 16);
      v16bf bk1 = *(const v16bf*)(krow + 32 + hv * 16);
#pragma unroll
      for (int qi = 0; qi < 2; ++qi) {
        s[qi][c] = wmma_bf16(aq[qi][0], bk0, s[qi][c]);
        s[qi][c] = wmma_bf16(aq[qi][1], bk1, s[qi][c]);
      }
    }
    // ---- mask + online softmax; P tiles -> LDS ----
#pragma unroll
    for (int qi = 0; qi < 2; ++qi) {
#pragma unroll
      for (int r = 0; r < 8; ++r) {
        int m = r + hv * 8;
        int jlim = M_LEN + q0 + qi * 16 + m;   // keys > jlim masked
        float s0 = s[qi][0][r], s1 = s[qi][1][r];
        if (kb + l16 > jlim)      s0 = -3.0e38f;
        if (kb + 16 + l16 > jlim) s1 = -3.0e38f;
        float vmax = fmaxf(s0, s1);
        vmax = fmaxf(vmax, __shfl_xor(vmax, 1));
        vmax = fmaxf(vmax, __shfl_xor(vmax, 2));
        vmax = fmaxf(vmax, __shfl_xor(vmax, 4));
        vmax = fmaxf(vmax, __shfl_xor(vmax, 8));
        float mo = mrow[qi][r];
        float mn = fmaxf(mo, vmax);
        mrow[qi][r] = mn;
        float corr = exp2f((mo - mn) * cs);
        float p0 = exp2f((s0 - mn) * cs);
        float p1 = exp2f((s1 - mn) * cs);
        float ps = p0 + p1;
        ps += __shfl_xor(ps, 1);
        ps += __shfl_xor(ps, 2);
        ps += __shfl_xor(ps, 4);
        ps += __shfl_xor(ps, 8);
        lrow[qi][r] = lrow[qi][r] * corr + ps;
#pragma unroll
        for (int j = 0; j < 4; ++j) o[qi][j][r] *= corr;
        plds[w][qi * 512 + m * 32 + l16]      = (__bf16)p0;
        plds[w][qi * 512 + m * 32 + 16 + l16] = (__bf16)p1;
      }
    }
    asm volatile("s_wait_dscnt 0" ::: "memory");
    // ---- re-read both P tiles in A-operand layout ----
    v16bf ap[2];
#pragma unroll
    for (int qi = 0; qi < 2; ++qi) {
      v8bf plo = *(const v8bf*)&plds[w][qi * 512 + l16 * 32 + hv * 16];
      v8bf phi = *(const v8bf*)&plds[w][qi * 512 + l16 * 32 + 8 + hv * 16];
      ap[qi] = __builtin_shufflevector(plo, phi, 0, 1, 2, 3, 4, 5, 6, 7,
                                       8, 9, 10, 11, 12, 13, 14, 15);
    }
    // ---- O += P @ V: V operands loaded once, used by both query tiles ----
#pragma unroll
    for (int j = 0; j < 4; ++j) {
      v16bf bv = *(const v16bf*)(vbase + (long)(j * 16 + l16) * T_LEN + kb +
                                 hv * 16);
      o[0][j] = wmma_bf16(ap[0], bv, o[0][j]);
      o[1][j] = wmma_bf16(ap[1], bv, o[1][j]);
    }
  }
  // ---- normalize and store bf16 vec [S*B][1024] ----
#pragma unroll
  for (int qi = 0; qi < 2; ++qi)
#pragma unroll
    for (int r = 0; r < 8; ++r) {
      float inv = 1.0f / lrow[qi][r];
      int m = r + hv * 8;
      long row = (long)(q0 + qi * 16 + m) * BATCH + b;
#pragma unroll
      for (int j = 0; j < 4; ++j)
        vec[row * DMODEL + h * DHEAD + j * 16 + l16] =
            (__bf16)(o[qi][j][r] * inv);
    }
}

// ---------------------------------------------------------------------------
// LayerNorm: one 256-thread block per row of 1024 fp32
// ---------------------------------------------------------------------------
__global__ void __launch_bounds__(256)
ln_kernel(const float* __restrict__ Y, const float* __restrict__ gamma,
          const float* __restrict__ beta, float* __restrict__ out) {
  __shared__ float red[16];
  int row = blockIdx.x;
  int tid = threadIdx.x;
  const float* yr = Y + (long)row * DMODEL;
  float4 v = ((const float4*)yr)[tid];
  float s = v.x + v.y + v.z + v.w;
  float q = v.x * v.x + v.y * v.y + v.z * v.z + v.w * v.w;
#pragma unroll
  for (int m = 1; m < 32; m <<= 1) {
    s += __shfl_xor(s, m);
    q += __shfl_xor(q, m);
  }
  int wv = tid >> 5, ln = tid & 31;
  if (ln == 0) { red[wv] = s; red[8 + wv] = q; }
  __syncthreads();
  if (wv == 0) {
    float ss = (ln < 8) ? red[ln] : 0.f;
    float qq = (ln < 8) ? red[8 + ln] : 0.f;
#pragma unroll
    for (int m = 1; m < 8; m <<= 1) {
      ss += __shfl_xor(ss, m);
      qq += __shfl_xor(qq, m);
    }
    if (ln == 0) { red[0] = ss; red[1] = qq; }
  }
  __syncthreads();
  float mu = red[0] * (1.0f / DMODEL);
  float var = red[1] * (1.0f / DMODEL) - mu * mu;
  float rs = rsqrtf(var + 1e-5f);
  float4 g = ((const float4*)gamma)[tid];
  float4 bt = ((const float4*)beta)[tid];
  float4 o;
  o.x = (v.x - mu) * rs * g.x + bt.x;
  o.y = (v.y - mu) * rs * g.y + bt.y;
  o.z = (v.z - mu) * rs * g.z + bt.z;
  o.w = (v.w - mu) * rs * g.w + bt.w;
  ((float4*)(out + (long)row * DMODEL))[tid] = o;
}

// ---------------------------------------------------------------------------
extern "C" void kernel_launch(void* const* d_in, const int* in_sizes, int n_in,
                              void* d_out, int out_size, void* d_ws,
                              size_t ws_size, hipStream_t stream) {
  const float* x     = (const float*)d_in[0];
  const float* pos   = (const float*)d_in[1];
  const float* mem   = (const float*)d_in[2];
  // d_in[3] = mask (computed analytically)
  const float* w_qkv = (const float*)d_in[4];
  const float* w_o   = (const float*)d_in[5];
  const float* gamma = (const float*)d_in[6];
  const float* beta  = (const float*)d_in[7];
  float* out = (float*)d_out;

  char* ws = (char*)d_ws;
  size_t off = 0;
  auto alloc = [&](size_t bytes) {
    char* p = ws + off;
    off += (bytes + 255) & ~(size_t)255;
    return p;
  };
  __bf16* c_bf  = (__bf16*)alloc((size_t)T_LEN * BATCH * DMODEL * 2);
  float*  xp    = (float*) alloc((size_t)S_LEN * BATCH * DMODEL * 4);
  __bf16* wqkvT = (__bf16*)alloc((size_t)E3 * DMODEL * 2);
  __bf16* woT   = (__bf16*)alloc((size_t)DMODEL * DMODEL * 2);
  __bf16* qkvb  = (__bf16*)alloc((size_t)T_LEN * BATCH * E3 * 2);
  __bf16* vtb   = (__bf16*)alloc((size_t)BATCH * NHEAD * DHEAD * T_LEN * 2);
  __bf16* vecb  = (__bf16*)alloc((size_t)S_LEN * BATCH * DMODEL * 2);
  float*  Yb    = (float*) alloc((size_t)S_LEN * BATCH * DMODEL * 4);

  prep_kernel<<<(T_LEN * BATCH * DMODEL + 255) / 256, 256, 0, stream>>>(
      x, pos, mem, xp, c_bf);
  tcast_kernel<<<(DMODEL * E3 + 255) / 256, 256, 0, stream>>>(
      w_qkv, wqkvT, DMODEL, E3);
  tcast_kernel<<<(DMODEL * DMODEL + 255) / 256, 256, 0, stream>>>(
      w_o, woT, DMODEL, DMODEL);
  qkv_gemm_kernel<<<dim3(64, 48), 128, 0, stream>>>(c_bf, wqkvT, qkvb);
  vtrans_kernel<<<(int)(((long)BATCH * NHEAD * DHEAD * T_LEN + 255) / 256),
                  256, 0, stream>>>(qkvb, vtb);
  attn_kernel<<<512, 128, 0, stream>>>(qkvb, vtb, vecb);
  out_gemm_kernel<<<dim3(32, 16), 128, 0, stream>>>(vecb, woT, xp, Yb);
  ln_kernel<<<S_LEN * BATCH, 256, 0, stream>>>(Yb, gamma, beta, out);
}
